// LSTMEncoder_83047487635608
// MI455X (gfx1250) — matrix operand order repaired and checked by reading, not compiled, not run on hardware
//
#include <hip/hip_runtime.h>
#include <hip/hip_bf16.h>

typedef __attribute__((ext_vector_type(16))) __bf16 v16bf;
typedef __attribute__((ext_vector_type(8)))  __bf16 v8bf;
typedef __attribute__((ext_vector_type(8)))  float  v8f;

// Global (address_space(1)) views of the bf16 weight array: guarantees the
// per-iteration laundered loads lower to global_load_b128 (LOADcnt only),
// not flat_load (which ties up DScnt and the LDS pipe too).
typedef __attribute__((address_space(1))) const unsigned short gu16;
typedef __attribute__((address_space(1))) const v16bf         gv16;

#define B_   256
#define T_   1024
#define H_   256
#define G4_  1024
#define HS_  264   // LDS row stride in bf16 elements (16B pad to stagger banks)

__device__ __forceinline__ unsigned short f2bf(float f) {
  union { float f; unsigned int u; } v; v.f = f;
  unsigned int u = v.u;
  u += 0x7fffu + ((u >> 16) & 1u);   // round-to-nearest-even
  return (unsigned short)(u >> 16);
}
__device__ __forceinline__ float bf2f(unsigned short b) {
  union { unsigned int u; float f; } v; v.u = ((unsigned int)b) << 16;
  return v.f;
}
__device__ __forceinline__ float sigmoid_(float x) {
  return 1.0f / (1.0f + __expf(-x));
}
__device__ __forceinline__ float tanh_(float x) {
  float e = __expf(2.0f * x);
  return (e - 1.0f) / (e + 1.0f);
}

// A fragment: 16x32 bf16 tile, row-major in LDS with stride HS_.
// lanes 0-15 -> row M=lane, K = k0..k0+15; lanes 16-31 -> same rows, K = k0+16..k0+31
// (contiguous order, matching load_b).
__device__ __forceinline__ v16bf load_a(const unsigned short* base, int lane, int k0) {
  const int row = lane & 15;
  const int kb  = k0 + ((lane >> 4) << 4);
  const __bf16* p = (const __bf16*)(base + row * HS_ + kb);
  v8bf lo = *(const v8bf*)(p);
  v8bf hi = *(const v8bf*)(p + 8);
  return __builtin_shufflevector(lo, hi, 0,1,2,3,4,5,6,7,8,9,10,11,12,13,14,15);
}

// B fragment: B[k][n] = W[nt+n][k] (gates = h @ W^T), W row-major [G4_][H_] bf16.
// contiguous k order (same as load_a): lanes 0-15 -> col N=lane, K=k0..k0+15; lanes 16-31 -> K=k0+16..k0+31.
__device__ __forceinline__ v16bf load_b(gu16* W, int nt, int lane, int k0) {
  const int n  = nt + (lane & 15);
  const int kb = k0 + ((lane >> 4) << 4);
  return *(gv16*)(W + n * H_ + kb);   // 32B contiguous, 32B aligned
}

__device__ __forceinline__ v8f wmma_bf16(v16bf a, v16bf b, v8f c) {
  return __builtin_amdgcn_wmma_f32_16x16x32_bf16(false, a, false, b, (short)0, c, false, false);
}

__global__ void prep_weights(const float* __restrict__ whh0,
                             const float* __restrict__ wih1,
                             const float* __restrict__ whh1,
                             unsigned short* __restrict__ wbf) {
  const int i = blockIdx.x * blockDim.x + threadIdx.x;
  if (i < G4_ * H_) {
    wbf[i]                = f2bf(whh0[i]);
    wbf[G4_ * H_ + i]     = f2bf(wih1[i]);
    wbf[2 * G4_ * H_ + i] = f2bf(whh1[i]);
  }
}

__global__ __launch_bounds__(512, 1)
void lstm2_persistent(const float* __restrict__ x,
                      const float* __restrict__ wih0,
                      const float* __restrict__ bih0, const float* __restrict__ bhh0,
                      const float* __restrict__ bih1, const float* __restrict__ bhh1,
                      const unsigned short* __restrict__ wbf,
                      float* __restrict__ out) {
  __shared__ unsigned short h0s[2][16][HS_];
  __shared__ unsigned short h1s[2][16][HS_];
  __shared__ float xs[16];

  const int tid  = threadIdx.x;
  const int wave = tid >> 5;        // 0..15, owns h-columns [wave*16, wave*16+16)
  const int lane = tid & 31;
  const int b0   = blockIdx.x * 16; // batch tile

  const int nI = wave * 16;         // gate tile row bases in the 4H dim (i,f,g,o)
  const int nF = 256 + nI;
  const int nG = 512 + nI;
  const int nO = 768 + nI;
  const int ncol = nI + (lane & 15);   // this lane's h column (D-frag N)
  const int mB   = (lane >> 4) << 3;   // D-frag M base: rows mB..mB+7

  // per-lane constants hoisted out of the time loop
  const float wxi = wih0[ncol];
  const float wxf = wih0[256 + ncol];
  const float wxg = wih0[512 + ncol];
  const float wxo = wih0[768 + ncol];
  const float bI0 = bih0[ncol]       + bhh0[ncol];
  const float bF0 = bih0[256 + ncol] + bhh0[256 + ncol];
  const float bG0 = bih0[512 + ncol] + bhh0[512 + ncol];
  const float bO0 = bih0[768 + ncol] + bhh0[768 + ncol];
  const float bI1 = bih1[ncol]       + bhh1[ncol];
  const float bF1 = bih1[256 + ncol] + bhh1[256 + ncol];
  const float bG1 = bih1[512 + ncol] + bhh1[512 + ncol];
  const float bO1 = bih1[768 + ncol] + bhh1[768 + ncol];

  float c0r[8], c1r[8];
  #pragma unroll
  for (int r = 0; r < 8; ++r) { c0r[r] = 0.0f; c1r[r] = 0.0f; }

  for (int i = tid; i < 2 * 16 * HS_; i += 512) {
    (&h0s[0][0][0])[i] = 0;
    (&h1s[0][0][0])[i] = 0;
  }
  __syncthreads();

  const v8f z = {0.f, 0.f, 0.f, 0.f, 0.f, 0.f, 0.f, 0.f};
  int buf = 0;

  // Opaque (laundered) weight base: redefined every iteration so the compiler
  // cannot hoist the 96 loop-invariant B-fragment loads out of the t-loop and
  // spill them to scratch. Cast back to an explicit global pointer so the
  // loads use the global (not flat) path.
  unsigned long long wlaunder = (unsigned long long)wbf;

  for (int t = 0; t < T_; ++t) {
    asm volatile("" : "+s"(wlaunder));
    gu16* Whh0 = (gu16*)wlaunder;
    gu16* Wih1 = Whh0 + G4_ * H_;
    gu16* Whh1 = Whh0 + 2 * G4_ * H_;

    if (tid < 16) xs[tid] = x[(b0 + tid) * T_ + t];
    __syncthreads();

    // ---------- layer 0: gates0 = h0 @ Whh0^T ----------
    const unsigned short* h0p = &h0s[buf][0][0];
    v8f aI = z, aF = z, aG = z, aO = z;
    for (int k = 0; k < H_; k += 32) {
      v16bf a = load_a(h0p, lane, k);
      aI = wmma_bf16(a, load_b(Whh0, nI, lane, k), aI);
      aF = wmma_bf16(a, load_b(Whh0, nF, lane, k), aF);
      aG = wmma_bf16(a, load_b(Whh0, nG, lane, k), aG);
      aO = wmma_bf16(a, load_b(Whh0, nO, lane, k), aO);
    }
    #pragma unroll
    for (int r = 0; r < 8; ++r) {
      const float xm = xs[mB + r];                       // IN==1: xp = x*W_ih0[n]
      const float gi = sigmoid_(aI[r] + bI0 + xm * wxi);
      const float gf = sigmoid_(aF[r] + bF0 + xm * wxf);
      const float gg = tanh_   (aG[r] + bG0 + xm * wxg);
      const float go = sigmoid_(aO[r] + bO0 + xm * wxo);
      const float c  = gf * c0r[r] + gi * gg;
      c0r[r] = c;
      h0s[buf ^ 1][mB + r][ncol] = f2bf(go * tanh_(c));
    }
    __syncthreads();

    // ---------- layer 1: gates1 = y0 @ Wih1^T + h1 @ Whh1^T ----------
    const unsigned short* y0p = &h0s[buf ^ 1][0][0];
    const unsigned short* h1p = &h1s[buf][0][0];
    v8f eI = z, eF = z, eG = z, eO = z;
    for (int k = 0; k < H_; k += 32) {
      v16bf ay = load_a(y0p, lane, k);
      eI = wmma_bf16(ay, load_b(Wih1, nI, lane, k), eI);
      eF = wmma_bf16(ay, load_b(Wih1, nF, lane, k), eF);
      eG = wmma_bf16(ay, load_b(Wih1, nG, lane, k), eG);
      eO = wmma_bf16(ay, load_b(Wih1, nO, lane, k), eO);
      v16bf ah = load_a(h1p, lane, k);
      eI = wmma_bf16(ah, load_b(Whh1, nI, lane, k), eI);
      eF = wmma_bf16(ah, load_b(Whh1, nF, lane, k), eF);
      eG = wmma_bf16(ah, load_b(Whh1, nG, lane, k), eG);
      eO = wmma_bf16(ah, load_b(Whh1, nO, lane, k), eO);
    }
    #pragma unroll
    for (int r = 0; r < 8; ++r) {
      const float gi = sigmoid_(eI[r] + bI1);
      const float gf = sigmoid_(eF[r] + bF1);
      const float gg = tanh_   (eG[r] + bG1);
      const float go = sigmoid_(eO[r] + bO1);
      const float c  = gf * c1r[r] + gi * gg;
      c1r[r] = c;
      h1s[buf ^ 1][mB + r][ncol] = f2bf(go * tanh_(c));
    }
    __syncthreads();
    buf ^= 1;
  }

  // ---------- outputs: h_n [2,B,H] then c_n [2,B,H] ----------
  for (int i = tid; i < 16 * H_; i += 512) {
    const int m = i >> 8, n = i & 255;
    out[(b0 + m) * H_ + n]             = bf2f(h0s[buf][m][n]);
    out[B_ * H_ + (b0 + m) * H_ + n]   = bf2f(h1s[buf][m][n]);
  }
  #pragma unroll
  for (int r = 0; r < 8; ++r) {
    const int m = mB + r;
    out[2 * B_ * H_ + (b0 + m) * H_ + ncol] = c0r[r];
    out[3 * B_ * H_ + (b0 + m) * H_ + ncol] = c1r[r];
  }
}

extern "C" void kernel_launch(void* const* d_in, const int* in_sizes, int n_in,
                              void* d_out, int out_size, void* d_ws, size_t ws_size,
                              hipStream_t stream) {
  (void)in_sizes; (void)n_in; (void)out_size; (void)ws_size;
  const float* x    = (const float*)d_in[0];
  const float* wih0 = (const float*)d_in[1];
  const float* whh0 = (const float*)d_in[2];
  const float* bih0 = (const float*)d_in[3];
  const float* bhh0 = (const float*)d_in[4];
  const float* wih1 = (const float*)d_in[5];
  const float* whh1 = (const float*)d_in[6];
  const float* bih1 = (const float*)d_in[7];
  const float* bhh1 = (const float*)d_in[8];
  unsigned short* wbf = (unsigned short*)d_ws;   // 3 * 1024*256 bf16 = 1.5 MB
  float* out = (float*)d_out;

  prep_weights<<<(G4_ * H_) / 256, 256, 0, stream>>>(whh0, wih1, whh1, wbf);
  lstm2_persistent<<<B_ / 16, 512, 0, stream>>>(x, wih0, bih0, bhh0, bih1, bhh1, wbf, out);
}
